// SingleLayerLstm_32074815766765
// MI455X (gfx1250) — matrix operand order repaired and checked by reading, not compiled, not run on hardware
//
#include <hip/hip_runtime.h>

// ---------------------------------------------------------------------------
// Single-layer LSTM for MI455X (gfx1250, wave32, WMMA, TDM).
//   T=2048, B=16, NIN=1024, NH=1024, G=4*NH=4096.
// Phase 1: intern = X @ W_in + b        (big parallel GEMM, bf16x2 WMMA)
// Phase 2: persistent scan kernel, W_re slice resident in 320KB LDS
//          (staged by the Tensor Data Mover), device-wide barrier per step,
//          h state double-buffered in L2.
// ---------------------------------------------------------------------------

#define T_STEPS 2048
#define BATCH   16
#define NIN     1024
#define NH      1024
#define GATES   4096          // 4*NH
#define KT_IN   (NIN/32)      // 32 k-tiles of 32
#define KT_RE   (NH/32)       // 32 k-tiles of 32
#define NT_G    (GATES/16)    // 256 n-tiles of 16
#define NBLK    64            // scan blocks; each owns 16 hidden columns

typedef __bf16    bf16x16 __attribute__((ext_vector_type(16)));
typedef float     f32x8   __attribute__((ext_vector_type(8)));
typedef float     f32x4   __attribute__((ext_vector_type(4)));
typedef unsigned  u32x4   __attribute__((ext_vector_type(4)));
typedef int       i32x4   __attribute__((ext_vector_type(4)));
typedef int       i32x8   __attribute__((ext_vector_type(8)));

union Frag {                 // one WMMA A/B operand (16 bf16 per lane)
    bf16x16        v;
    u32x4          q[2];
    unsigned short u[16];
};

// native gfx1250 bf16 converts (v_cvt_*bf16*), RNE
__device__ __forceinline__ unsigned short f2bf(float f) {
    __bf16 b = (__bf16)f;
    return __builtin_bit_cast(unsigned short, b);
}
__device__ __forceinline__ float bf2f(unsigned short h) {
    return (float)__builtin_bit_cast(__bf16, h);
}

// D = A*B + C with fp32-ish accuracy from bf16 hi/lo split:
//   A*B ~= Ah*Bh + Ah*Bl + Al*Bh   (accumulated in f32 C)
#define WMMA_BF16(A, B, C) \
    __builtin_amdgcn_wmma_f32_16x16x32_bf16(false, (A), false, (B), (short)0, (C), false, false)

// --- packed fragment index helpers (per CDNA5 ISA 7.12.2 VGPR layouts) ------
// A-matrix 16x32 bf16: lane = (K-group)*16 + M ; 16 slots/lane:
//   s<8  -> K = half*8 + s          (half = lane>>4)
//   s>=8 -> K = 16 + half*8 + (s-8)
// B-matrix 32x16 bf16: lane = half*16 + N ; slot s -> K = (s>>3)*16 + half*8 + (s&7).
// Tiles stored contiguously: PIDX(nt,kt) = (nt*KT + kt)*512 + lane*16 + s.

// ---------------------------------------------------------------------------
// init: zero grid-barrier counter; scatter h0 into packed-A layout (buffer 0)
// ---------------------------------------------------------------------------
__global__ void init_kernel(unsigned* __restrict__ cnt,
                            const float* __restrict__ h0,
                            unsigned short* __restrict__ hph,
                            unsigned short* __restrict__ hpl) {
    int tid = blockIdx.x * blockDim.x + threadIdx.x;
    if (tid == 0) *cnt = 0u;
    if (tid < BATCH * NH) {
        int m = tid >> 10;          // batch row
        int k = tid & 1023;         // hidden index
        int kt = k >> 5, kk = k & 31;
        int half, s;
        if (kk < 16) { half = kk >> 3; s = kk & 7; }
        else         { half = (kk - 16) >> 3; s = 8 + ((kk - 16) & 7); }
        int idx = kt * 512 + (half * 16 + m) * 16 + s;
        float f = h0[tid];
        unsigned short hb = f2bf(f);
        hph[idx] = hb;
        hpl[idx] = f2bf(f - bf2f(hb));
    }
}

// ---------------------------------------------------------------------------
// pack a 1024x4096 f32 weight matrix into WMMA B-fragment order, bf16 hi+lo
// ---------------------------------------------------------------------------
__global__ __launch_bounds__(256)
void pack_w_kernel(const float* __restrict__ W,
                   unsigned short* __restrict__ hi,
                   unsigned short* __restrict__ lo) {
    int tid = blockIdx.x * 256 + threadIdx.x;     // == PIDX by construction
    if (tid >= NT_G * KT_IN * 512) return;
    int s    =  tid        & 15;
    int lane = (tid >> 4)  & 31;
    int kt   = (tid >> 9)  & 31;
    int nt   =  tid >> 14;
    int k = kt * 32 + ((s & 8) << 1) + (lane >> 4) * 8 + (s & 7);
    int n = nt * 16 + (lane & 15);
    float f = W[(size_t)k * GATES + n];
    unsigned short hb = f2bf(f);
    hi[tid] = hb;
    lo[tid] = f2bf(f - bf2f(hb));
}

// ---------------------------------------------------------------------------
// Phase 1: intern[M=32768][G=4096] = X @ W_in + b   (bf16x2 WMMA)
// grid = (2048 M-tiles, 32), block = 256 threads = 8 waves, wave -> one n-tile
// ---------------------------------------------------------------------------
__global__ __launch_bounds__(256)
void intern_gemm_kernel(const float* __restrict__ X,
                        const unsigned short* __restrict__ winh,
                        const unsigned short* __restrict__ winl,
                        const float* __restrict__ bias,
                        float* __restrict__ intern) {
    const int mt   = blockIdx.x;
    const int wv   = threadIdx.x >> 5;
    const int lane = threadIdx.x & 31;
    const int nt   = blockIdx.y * 8 + wv;

    const int m    = lane & 15;
    const int half = lane >> 4;
    const float* xrow = X + (size_t)(mt * 16 + m) * NIN;
    const u32x4* bh4 = (const u32x4*)winh;
    const u32x4* bl4 = (const u32x4*)winl;

    f32x8 acc = {0.f, 0.f, 0.f, 0.f, 0.f, 0.f, 0.f, 0.f};

    for (int kt = 0; kt < KT_IN; ++kt) {
        // A fragment: convert X f32 -> bf16 hi/lo on the fly with native
        // v_cvt converts (same HBM bytes as loading pre-split bf16 pairs)
        const float* xa = xrow + kt * 32 + half * 8;        // slots 0..7
        const float* xb = xa + 16;                          // slots 8..15
        f32x4 x0 = *(const f32x4*)xa;
        f32x4 x1 = *(const f32x4*)(xa + 4);
        f32x4 x2 = *(const f32x4*)xb;
        f32x4 x3 = *(const f32x4*)(xb + 4);
        Frag ah, al;
#pragma unroll
        for (int s = 0; s < 16; ++s) {
            float f = (s < 4) ? x0[s] : (s < 8) ? x1[s - 4]
                    : (s < 12) ? x2[s - 8] : x3[s - 12];
            __bf16 hb = (__bf16)f;
            ah.v[s] = hb;
            al.v[s] = (__bf16)(f - (float)hb);
        }
        // B fragment: two contiguous b128 loads per lane from packed W_in
        Frag bh, bl;
        int fb = (nt * KT_IN + kt) * 64 + lane * 2;
        bh.q[0] = bh4[fb]; bh.q[1] = bh4[fb + 1];
        bl.q[0] = bl4[fb]; bl.q[1] = bl4[fb + 1];

        acc = WMMA_BF16(ah.v, bl.v, acc);
        acc = WMMA_BF16(al.v, bh.v, acc);
        acc = WMMA_BF16(ah.v, bh.v, acc);
    }

    // C layout: VGPR r, lane l -> (m = r + (l>>4)*8, n = l&15)
    const int n  = nt * 16 + (lane & 15);
    const float bv = bias[n];
    const int m0 = mt * 16 + (lane >> 4) * 8;
#pragma unroll
    for (int r = 0; r < 8; ++r)
        intern[(size_t)(m0 + r) * GATES + n] = acc[r] + bv;
}

// ---------------------------------------------------------------------------
// Phase 2: persistent recurrent scan.
// 64 blocks x 256 threads (8 waves). Block bb owns hidden cols [16bb,16bb+16),
// i.e. gate n-tiles {q*64+bb : q=0..3}. Its W_re slice (4 gates x 32 ktiles,
// bf16 hi+lo = 256 KB) is staged into LDS once by the Tensor Data Mover and
// reused for all 2048 steps.
// Waves: (q = w>>1) gate tile, (kh = w&1) half of K -> 16 ktiles * 3 WMMA.
// Partials merged into LDS z-buffer with ds_add_f32 atomics.
// ---------------------------------------------------------------------------
__global__ __launch_bounds__(256)
void lstm_scan_kernel(const float* __restrict__ intern,
                      const float* __restrict__ imask,
                      const float* __restrict__ c0,
                      unsigned short* __restrict__ hph0, unsigned short* __restrict__ hpl0,
                      unsigned short* __restrict__ hph1, unsigned short* __restrict__ hpl1,
                      const unsigned short* __restrict__ wreh,
                      const unsigned short* __restrict__ wrel,
                      float* __restrict__ Y, float* __restrict__ C,
                      float* __restrict__ Dfin,
                      unsigned* __restrict__ cnt) {
    extern __shared__ u32x4 smem4[];            // 267264 B (< 320 KB WGP LDS)
    u32x4* wlds = smem4;                        // [16384] : W hi (8192) | W lo (8192)
    float* zb   = (float*)(smem4 + 16384);      // [4][16][16] gate pre-activations
    float* cb   = zb + 4 * 256;                 // [16][16] cell state

    const int bb   = blockIdx.x;
    const int tid  = threadIdx.x;
    const int wv   = tid >> 5, lane = tid & 31;
    const int q    = wv >> 1, kh = wv & 1;      // gate tile / K half
    const int m    = tid >> 4, j = tid & 15;    // gate-math mapping
    const int col  = bb * 16 + j;

    // ---- stage W_re slice into LDS with the TDM (once; reused 2048 steps).
    // 2D tile: row = one gate's 32 packed k-tiles (32 KB contiguous), 4 rows
    // strided by the 64-n-tile gate pitch. data_size=8B units.
    if (wv == 0) {
        const unsigned ROWU = 32 * 512 * 2 / 8;           // 4096 8B units / row
        const unsigned STRU = 64u * 32u * 512u * 2u / 8u; // 262144 units gate pitch
        i32x4 gz4 = {0, 0, 0, 0};
        i32x8 gz8 = {0, 0, 0, 0, 0, 0, 0, 0};
        i32x8 g1 = {(int)(3u << 16),                 // data_size=3 (8B)
                    (int)((ROWU & 0xFFFFu) << 16),   // tensor_dim0 lo16
                    (int)((ROWU >> 16) | (4u << 16)),// tensor_dim0 hi16 | tensor_dim1 lo16
                    (int)((ROWU & 0xFFFFu) << 16),   // tensor_dim1 hi16(0) | tile_dim0
                    (int)4,                          // tile_dim1=4, tile_dim2=0
                    (int)STRU,                       // tensor_dim0_stride lo32
                    0, 0};                           // stride hi / dim1_stride
        // hi fragments -> wlds[0 .. 8191]
        {
            unsigned long long ga =
                (unsigned long long)(uintptr_t)(wreh + (size_t)bb * 32 * 512);
            unsigned lds = (unsigned)(uintptr_t)wlds;
            u32x4 g0 = {1u, lds, (unsigned)ga,
                        (unsigned)((ga >> 32) & 0x1FFFFFFu) | (2u << 30)};
            __builtin_amdgcn_tensor_load_to_lds(g0, g1, gz4, gz4, gz8, 0);
        }
        // lo fragments -> wlds[8192 .. 16383]
        {
            unsigned long long ga =
                (unsigned long long)(uintptr_t)(wrel + (size_t)bb * 32 * 512);
            unsigned lds = (unsigned)(uintptr_t)(wlds + 8192);
            u32x4 g0 = {1u, lds, (unsigned)ga,
                        (unsigned)((ga >> 32) & 0x1FFFFFFu) | (2u << 30)};
            __builtin_amdgcn_tensor_load_to_lds(g0, g1, gz4, gz4, gz8, 0);
        }
        __builtin_amdgcn_s_wait_tensorcnt(0);
    }
    cb[m * 16 + j] = c0[m * NH + col];
    __syncthreads();

    unsigned target = 0;
    for (int t = 0; t < T_STEPS; ++t) {
        // double-buffered packed h state (read t&1==0 -> buf0)
        const u32x4* ah4 = (const u32x4*)((t & 1) ? hph1 : hph0);
        const u32x4* al4 = (const u32x4*)((t & 1) ? hpl1 : hpl0);
        unsigned short* wh = (t & 1) ? hph0 : hph1;
        unsigned short* wl = (t & 1) ? hpl0 : hpl1;

        // ---- z := intern[t] slice (bias already folded in) ----
        for (int i = tid; i < 1024; i += 256) {
            int qq = i >> 8, mm = (i >> 4) & 15, jj = i & 15;
            zb[i] = intern[(size_t)(t * 16 + mm) * GATES + qq * NH + bb * 16 + jj];
        }
        __syncthreads();

        // ---- z += h @ W_re  (bf16x2 WMMA; B from LDS, A from L2) ----
        f32x8 acc = {0.f, 0.f, 0.f, 0.f, 0.f, 0.f, 0.f, 0.f};
        for (int kt = kh * 16; kt < kh * 16 + 16; ++kt) {
            Frag ah, al, bh, bl;
            int ab = kt * 64 + lane * 2;
            ah.q[0] = ah4[ab]; ah.q[1] = ah4[ab + 1];
            al.q[0] = al4[ab]; al.q[1] = al4[ab + 1];
            int fb = (q * KT_RE + kt) * 64 + lane * 2;
            bh.q[0] = wlds[fb];        bh.q[1] = wlds[fb + 1];
            bl.q[0] = wlds[8192 + fb]; bl.q[1] = wlds[8192 + fb + 1];
            acc = WMMA_BF16(ah.v, bl.v, acc);
            acc = WMMA_BF16(al.v, bh.v, acc);
            acc = WMMA_BF16(ah.v, bh.v, acc);
        }
        {
            int n = lane & 15, mb = (lane >> 4) * 8;
#pragma unroll
            for (int r = 0; r < 8; ++r)
                atomicAdd(&zb[q * 256 + (mb + r) * 16 + n], acc[r]);  // ds_add_f32
        }
        // prefetch next step's intern slice while z is being merged
        if (t + 1 < T_STEPS)
            __builtin_prefetch(&intern[(size_t)((t + 1) * 16 + m) * GATES + q * NH + bb * 16 + j], 0, 1);
        __syncthreads();

        // ---- gate nonlinearity + state update (one thread per (m,j)) ----
        float zc = zb[0 * 256 + m * 16 + j];
        float zi = zb[1 * 256 + m * 16 + j];
        float zf = zb[2 * 256 + m * 16 + j];
        float zo = zb[3 * 256 + m * 16 + j];
        float e2 = __expf(-2.f * zc);
        float ci = (1.f - e2) / (1.f + e2);                  // tanh
        float ig = 1.f / (1.f + __expf(-zi));
        float fg = 1.f / (1.f + __expf(-zf));
        float og = 1.f / (1.f + __expf(-zo));
        float cold = cb[m * 16 + j];
        float cn = fg * cold + ig * ci;
        float ec = __expf(-2.f * cn);
        float yn = og * (1.f - ec) / (1.f + ec);
        float mk = imask[t * BATCH + m];
        float ck = mk * cn + (1.f - mk) * cold;
        float yv = mk * yn;
        cb[m * 16 + j] = ck;

        size_t o = (size_t)t * (BATCH * NH) + (size_t)m * NH + col;
        Y[o] = yv;
        C[o] = ck;
        if (t == T_STEPS - 1) Dfin[m * NH + col] = ck;

        // ---- publish h=y into next packed-A buffer (bf16 hi/lo) ----
        {
            int k = col, kt = k >> 5, kk = k & 31;
            int half, s;
            if (kk < 16) { half = kk >> 3; s = kk & 7; }
            else         { half = (kk - 16) >> 3; s = 8 + ((kk - 16) & 7); }
            int idx = kt * 512 + (half * 16 + m) * 16 + s;
            unsigned short hb = f2bf(yv);
            wh[idx] = hb;
            wl[idx] = f2bf(yv - bf2f(hb));
        }

        // ---- device-wide barrier (all blocks' h writes visible) ----
        __threadfence();
        __syncthreads();                 // whole block done before arriving
        target += NBLK;
        if (tid == 0) {
            __threadfence();
            atomicAdd(cnt, 1u);
            while (__hip_atomic_load(cnt, __ATOMIC_ACQUIRE,
                                     __HIP_MEMORY_SCOPE_AGENT) < target)
                __builtin_amdgcn_s_sleep(1);
        }
        __syncthreads();
    }
}

// ---------------------------------------------------------------------------
extern "C" void kernel_launch(void* const* d_in, const int* in_sizes, int n_in,
                              void* d_out, int out_size, void* d_ws, size_t ws_size,
                              hipStream_t stream) {
    const float* X    = (const float*)d_in[0];   // (T,B,NIN)
    const float* imsk = (const float*)d_in[1];   // (T,B)
    const float* h0   = (const float*)d_in[2];   // (B,NH)
    const float* c0   = (const float*)d_in[3];   // (B,NH)
    const float* W_in = (const float*)d_in[4];   // (NIN,G)
    const float* W_re = (const float*)d_in[5];   // (NH,G)
    const float* bias = (const float*)d_in[6];   // (G,)

    // workspace layout (all chunks 256B-aligned)
    char* ws = (char*)d_ws;
    size_t off = 0;
    unsigned* cnt = (unsigned*)(ws + off);              off += 256;
    unsigned short* hph0 = (unsigned short*)(ws + off); off += (size_t)BATCH * NH * 2;
    unsigned short* hpl0 = (unsigned short*)(ws + off); off += (size_t)BATCH * NH * 2;
    unsigned short* hph1 = (unsigned short*)(ws + off); off += (size_t)BATCH * NH * 2;
    unsigned short* hpl1 = (unsigned short*)(ws + off); off += (size_t)BATCH * NH * 2;
    const size_t wpack = (size_t)NT_G * KT_IN * 512;    // packed weight elems
    unsigned short* winh = (unsigned short*)(ws + off); off += wpack * 2;
    unsigned short* winl = (unsigned short*)(ws + off); off += wpack * 2;
    unsigned short* wreh = (unsigned short*)(ws + off); off += wpack * 2;
    unsigned short* wrel = (unsigned short*)(ws + off); off += wpack * 2;
    float* intern = (float*)(ws + off);                 off += (size_t)T_STEPS * BATCH * GATES * 4;

    float* Y = (float*)d_out;
    float* C = Y + (size_t)T_STEPS * BATCH * NH;
    float* D = C + (size_t)T_STEPS * BATCH * NH;

    init_kernel<<<(BATCH * NH + 255) / 256, 256, 0, stream>>>(cnt, h0, hph0, hpl0);
    pack_w_kernel<<<(int)(wpack / 256), 256, 0, stream>>>(W_in, winh, winl);
    pack_w_kernel<<<(int)(wpack / 256), 256, 0, stream>>>(W_re, wreh, wrel);
    intern_gemm_kernel<<<dim3(T_STEPS * BATCH / 16, NT_G / 8), 256, 0, stream>>>(
        X, winh, winl, bias, intern);
    const size_t smem = 16384 * sizeof(u32x4) + 4 * 256 * 4 + 256 * 4; // 267264 B
    lstm_scan_kernel<<<NBLK, 256, smem, stream>>>(
        intern, imsk, c0, hph0, hpl0, hph1, hpl1, wreh, wrel, Y, C, D, cnt);
}
